// DecoderLayer_78460462564044
// MI455X (gfx1250) — compile-verified, matrix operand order checked by reading, not run on hardware
//
#include <hip/hip_runtime.h>
#include <hip/hip_bf16.h>

// ---------------------------------------------------------------------------
// Transformer decoder layer for MI455X (gfx1250, wave32, WMMA).
// All GEMMs + attention matmuls run on v_wmma_f32_16x16x32_f16.
// Contiguous LDS staging uses GLOBAL_LOAD_ASYNC_TO_LDS_B128 when available.
// ---------------------------------------------------------------------------

typedef _Float16 v16h __attribute__((ext_vector_type(16)));
typedef float    v8f  __attribute__((ext_vector_type(8)));
typedef int      v4i_vs __attribute__((vector_size(16)));   // b128 payload type

#define D_MODEL 512
#define HEADS   8
#define DK      64

#if __has_builtin(__builtin_amdgcn_global_load_async_to_lds_b128) && \
    __has_builtin(__builtin_amdgcn_s_wait_asynccnt)
#define HAVE_ASYNC_LDS 1
__device__ __forceinline__ void async_b128(_Float16* lds, const _Float16* g) {
  __builtin_amdgcn_global_load_async_to_lds_b128(
      (v4i_vs*)g, (v4i_vs*)lds, 0, 0);
}
__device__ __forceinline__ void async_wait0() {
  __builtin_amdgcn_s_wait_asynccnt(0);
}
#else
#define HAVE_ASYNC_LDS 0
#endif

// Per-lane WMMA fragment load: halves 0..7 at K=kA.., halves 8..15 at K=kA+16..
// (kA = 0 for lanes 0-15, 8 for lanes 16-31) -> two b128 loads.
__device__ __forceinline__ v16h frag_ld(const _Float16* p, int kA) {
  union { v16h v; uint4 u[2]; } f;
  f.u[0] = *(const uint4*)(p + kA);
  f.u[1] = *(const uint4*)(p + kA + 16);
  return f.v;
}

__device__ __forceinline__ v8f wmma16(v16h a, v16h b, v8f c) {
  return __builtin_amdgcn_wmma_f32_16x16x32_f16(false, a, false, b, (short)0, c,
                                                false, false);
}

// ---------------------------------------------------------------------------
// fp32 -> fp16 conversion (activations + weights)
// ---------------------------------------------------------------------------
__global__ void f32_to_f16_kernel(const float* __restrict__ a,
                                  _Float16* __restrict__ o, int n) {
  int i = blockIdx.x * blockDim.x + threadIdx.x;
  int stride = gridDim.x * blockDim.x;
  for (; i < n; i += stride) o[i] = (_Float16)a[i];
}

// ---------------------------------------------------------------------------
// WMMA GEMM:  C[M,N] = A[M,K] (f16) * W[K,N] (f16) + bias, optional relu,
// optional residual add (f32), writes f32 and/or f16 outputs.
// Block: 256 threads (8 waves). Block tile 128x64; wave tile 32x32 (2x2 WMMA).
// ---------------------------------------------------------------------------
__global__ __launch_bounds__(256) void gemm_wmma_kernel(
    const _Float16* __restrict__ A, const _Float16* __restrict__ W,
    const float* __restrict__ bias, const float* __restrict__ resid,
    float* __restrict__ Cf, _Float16* __restrict__ Ch,
    int M, int N, int K, int do_relu)
{
  __shared__ _Float16 As[128 * 40];   // 128 rows x 32 K (+8 pad)
  __shared__ _Float16 Bst[64 * 40];   // 64 cols x 32 K (+8 pad), transposed

  const int tid  = threadIdx.x;
  const int lane = tid & 31;
  const int wave = tid >> 5;
  const int bm = blockIdx.y * 128;
  const int bn = blockIdx.x * 64;
  const int m0 = (wave & 3) * 32;
  const int n0 = (wave >> 2) * 32;
  const int kA  = (lane & 16) ? 8 : 0;
  const int l16 = lane & 15;

  v8f c00 = {}, c01 = {}, c10 = {}, c11 = {};

  const int arow = tid >> 1, aseg = (tid & 1) * 16;  // A: 2 thr/row, 16 halves
  const int brow = tid >> 3, bcol = (tid & 7) * 8;   // W: 8 thr/row, 8 halves

  for (int kt = 0; kt < K; kt += 32) {
    __syncthreads();
    {
      const _Float16* ga = A + (size_t)(bm + arow) * K + kt + aseg;
#if HAVE_ASYNC_LDS
      async_b128(As + arow * 40 + aseg,     ga);
      async_b128(As + arow * 40 + aseg + 8, ga + 8);
#else
      uint4 a0 = *(const uint4*)ga;
      uint4 a1 = *(const uint4*)(ga + 8);
      *(uint4*)(As + arow * 40 + aseg)     = a0;
      *(uint4*)(As + arow * 40 + aseg + 8) = a1;
#endif
      union { uint4 u; _Float16 h[8]; } wv;
      wv.u = *(const uint4*)(W + (size_t)(kt + brow) * N + bn + bcol);
      #pragma unroll
      for (int i = 0; i < 8; ++i) Bst[(bcol + i) * 40 + brow] = wv.h[i];
      if (kt + 32 < K) {
        __builtin_prefetch(A + (size_t)(bm + arow) * K + kt + 32 + aseg, 0, 0);
        __builtin_prefetch(W + (size_t)(kt + 32 + brow) * N + bn + bcol, 0, 0);
      }
#if HAVE_ASYNC_LDS
      async_wait0();   // own async copies landed in LDS before barrier signal
#endif
    }
    __syncthreads();

    v16h a0f = frag_ld(As + (m0 + l16) * 40, kA);
    v16h a1f = frag_ld(As + (m0 + 16 + l16) * 40, kA);
    v16h b0f = frag_ld(Bst + (n0 + l16) * 40, kA);
    v16h b1f = frag_ld(Bst + (n0 + 16 + l16) * 40, kA);
    c00 = wmma16(a0f, b0f, c00);
    c01 = wmma16(a0f, b1f, c01);
    c10 = wmma16(a1f, b0f, c10);
    c11 = wmma16(a1f, b1f, c11);
  }

  const int rhi = (lane >> 4) << 3;   // C layout: VGPR r -> rows r / r+8
  #pragma unroll
  for (int mt = 0; mt < 2; ++mt) {
    #pragma unroll
    for (int nt = 0; nt < 2; ++nt) {
      v8f c = mt ? (nt ? c11 : c10) : (nt ? c01 : c00);
      const int gcol = bn + n0 + nt * 16 + l16;
      const float bv = bias[gcol];
      #pragma unroll
      for (int r = 0; r < 8; ++r) {
        const int grow = bm + m0 + mt * 16 + r + rhi;
        float v = c[r] + bv;
        if (do_relu) v = fmaxf(v, 0.f);
        if (resid) v += resid[(size_t)grow * N + gcol];
        if (Cf) Cf[(size_t)grow * N + gcol] = v;
        if (Ch) Ch[(size_t)grow * N + gcol] = (_Float16)v;
      }
    }
  }
}

// ---------------------------------------------------------------------------
// Flash attention (per block: one (b, h, 64-query tile); 4 waves x 16 q rows).
// S = QK^T * scale (+mask), online softmax, O = P V, write f16 [B*L, D].
// ---------------------------------------------------------------------------
__global__ __launch_bounds__(128) void attn_wmma_kernel(
    const _Float16* __restrict__ Q, const _Float16* __restrict__ Kmat,
    const _Float16* __restrict__ V, _Float16* __restrict__ O,
    const int* __restrict__ mask, int L, int maskB, int maskQ, float scale)
{
  __shared__ _Float16 Ks[32 * 72];      // 32 keys x 64 dk (+8 pad), row-major
  __shared__ _Float16 Vt[64 * 40];      // 64 dk x 32 keys (+8 pad), transposed
  __shared__ _Float16 Pb[4 * 16 * 40];  // wave-private P staging (16x32 +pad)

  const int tid  = threadIdx.x;
  const int lane = tid & 31;
  const int wave = tid >> 5;
  const int qtiles = L >> 6;
  const int qt = blockIdx.x % qtiles;
  const int h  = (blockIdx.x / qtiles) % HEADS;
  const int b  = blockIdx.x / (qtiles * HEADS);
  const int qbase = qt * 64 + wave * 16;
  const int kA  = (lane & 16) ? 8 : 0;
  const int l16 = lane & 15;
  const int rhi = (lane >> 4) << 3;

  const _Float16* qrow = Q + (size_t)(b * L + qbase + l16) * D_MODEL + h * DK;
  v16h qf0 = frag_ld(qrow, kA);        // dk 0..31
  v16h qf1 = frag_ld(qrow + 32, kA);   // dk 32..63

  v8f o0 = {}, o1 = {}, o2 = {}, o3 = {};
  float mr[8], lr[8];
  #pragma unroll
  for (int r = 0; r < 8; ++r) { mr[r] = -1e30f; lr[r] = 0.f; }

  const int krow = tid >> 2, seg = (tid & 3) * 16;
  const int* mb = mask + (size_t)b * maskB;
  _Float16* pw = Pb + wave * 16 * 40;

  for (int kb = 0; kb < L; kb += 32) {
    __syncthreads();
    {
      const size_t src = (size_t)(b * L + kb + krow) * D_MODEL + h * DK + seg;
#if HAVE_ASYNC_LDS
      async_b128(Ks + krow * 72 + seg,     Kmat + src);
      async_b128(Ks + krow * 72 + seg + 8, Kmat + src + 8);
#else
      uint4 k0 = *(const uint4*)(Kmat + src);
      uint4 k1 = *(const uint4*)(Kmat + src + 8);
      *(uint4*)(Ks + krow * 72 + seg)     = k0;
      *(uint4*)(Ks + krow * 72 + seg + 8) = k1;
#endif
      union { uint4 u; _Float16 hh[8]; } v0u, v1u;
      v0u.u = *(const uint4*)(V + src);
      v1u.u = *(const uint4*)(V + src + 8);
      #pragma unroll
      for (int i = 0; i < 8; ++i) {
        Vt[(seg + i) * 40 + krow]     = v0u.hh[i];
        Vt[(seg + 8 + i) * 40 + krow] = v1u.hh[i];
      }
#if HAVE_ASYNC_LDS
      async_wait0();
#endif
    }
    __syncthreads();

    // scores: two 16x16 tiles over this 32-key chunk, contracting dk=64
    v8f s0 = {}, s1 = {};
    s0 = wmma16(qf0, frag_ld(Ks + l16 * 72, kA), s0);
    s0 = wmma16(qf1, frag_ld(Ks + l16 * 72 + 32, kA), s0);
    s1 = wmma16(qf0, frag_ld(Ks + (16 + l16) * 72, kA), s1);
    s1 = wmma16(qf1, frag_ld(Ks + (16 + l16) * 72 + 32, kA), s1);

    const int c0 = kb + l16, c1 = c0 + 16;
    float al[8];
    #pragma unroll
    for (int r = 0; r < 8; ++r) {
      const int qr = qbase + r + rhi;
      const int* mp = mb + (size_t)qr * maskQ;   // maskQ==0 broadcasts row
      float v0 = s0[r] * scale; if (mp[c0] == 0) v0 = -1e9f;
      float v1 = s1[r] * scale; if (mp[c1] == 0) v1 = -1e9f;
      float mx = fmaxf(v0, v1);
      #pragma unroll
      for (int off = 8; off; off >>= 1) mx = fmaxf(mx, __shfl_xor(mx, off, 32));
      const float mn = fmaxf(mr[r], mx);
      const float a  = __expf(mr[r] - mn);
      mr[r] = mn;
      v0 = __expf(v0 - mn);
      v1 = __expf(v1 - mn);
      float rs = v0 + v1;
      #pragma unroll
      for (int off = 8; off; off >>= 1) rs += __shfl_xor(rs, off, 32);
      lr[r] = lr[r] * a + rs;
      al[r] = a;
      // stash P (C-layout -> LDS) for re-load in A-layout
      pw[(r + rhi) * 40 + l16]      = (_Float16)v0;
      pw[(r + rhi) * 40 + 16 + l16] = (_Float16)v1;
    }
    #pragma unroll
    for (int r = 0; r < 8; ++r) {
      o0[r] *= al[r]; o1[r] *= al[r]; o2[r] *= al[r]; o3[r] *= al[r];
    }
    v16h pf = frag_ld(pw + l16 * 40, kA);
    o0 = wmma16(pf, frag_ld(Vt + l16 * 40, kA), o0);
    o1 = wmma16(pf, frag_ld(Vt + (16 + l16) * 40, kA), o1);
    o2 = wmma16(pf, frag_ld(Vt + (32 + l16) * 40, kA), o2);
    o3 = wmma16(pf, frag_ld(Vt + (48 + l16) * 40, kA), o3);
  }

  #pragma unroll
  for (int r = 0; r < 8; ++r) {
    const float inv = lr[r] > 0.f ? 1.f / lr[r] : 0.f;
    _Float16* op = O + (size_t)(b * L + qbase + r + rhi) * D_MODEL + h * DK + l16;
    op[0]  = (_Float16)(o0[r] * inv);
    op[16] = (_Float16)(o1[r] * inv);
    op[32] = (_Float16)(o2[r] * inv);
    op[48] = (_Float16)(o3[r] * inv);
  }
}

// ---------------------------------------------------------------------------
// LayerNorm (torch semantics: unbiased std, eps added to std), one wave/row.
// Emits f32 (residual chain) and optional f16 (next GEMM input).
// ---------------------------------------------------------------------------
__global__ __launch_bounds__(256) void layernorm_kernel(
    const float* __restrict__ X, const float* __restrict__ alpha,
    const float* __restrict__ beta, float* __restrict__ Yf,
    _Float16* __restrict__ Yh, int rows)
{
  const int lane = threadIdx.x & 31;
  const int row = blockIdx.x * 8 + (threadIdx.x >> 5);
  if (row >= rows) return;
  const float* xp = X + (size_t)row * D_MODEL;
  float xv[16], s = 0.f, q = 0.f;
  #pragma unroll
  for (int i = 0; i < 16; ++i) {
    xv[i] = xp[i * 32 + lane];
    s += xv[i];
    q += xv[i] * xv[i];
  }
  #pragma unroll
  for (int off = 16; off; off >>= 1) {
    s += __shfl_xor(s, off, 32);
    q += __shfl_xor(q, off, 32);
  }
  const float mean = s * (1.f / D_MODEL);
  float var = (q - (float)D_MODEL * mean * mean) * (1.f / (D_MODEL - 1));
  var = fmaxf(var, 0.f);
  const float inv = 1.f / (sqrtf(var) + 1e-6f);
  #pragma unroll
  for (int i = 0; i < 16; ++i) {
    const int col = i * 32 + lane;
    const float y = alpha[col] * (xv[i] - mean) * inv + beta[col];
    Yf[(size_t)row * D_MODEL + col] = y;
    if (Yh) Yh[(size_t)row * D_MODEL + col] = (_Float16)y;
  }
}

// ---------------------------------------------------------------------------
// Host orchestration
// ---------------------------------------------------------------------------
extern "C" void kernel_launch(void* const* d_in, const int* in_sizes, int n_in,
                              void* d_out, int out_size, void* d_ws, size_t ws_size,
                              hipStream_t stream) {
  (void)n_in; (void)out_size; (void)ws_size;

  const float* x = (const float*)d_in[0];
  const float* e = (const float*)d_in[1];
  const int* src_mask = (const int*)d_in[2];
  const int* trg_mask = (const int*)d_in[3];

  const int R = in_sizes[0] / D_MODEL;      // B*L
  const int L = in_sizes[3] / in_sizes[2];  // (B*L*L)/(B*L)
  const int B = R / L;
  const int F = in_sizes[20] / D_MODEL;

  size_t cur = 0;
  char* wsb = (char*)d_ws;
  auto alloc = [&](size_t bytes) -> void* {
    void* p = wsb + cur;
    cur += (bytes + 255) & ~(size_t)255;
    return p;
  };

  const size_t act = (size_t)R * D_MODEL;
  _Float16* xh  = (_Float16*)alloc(act * 2);
  _Float16* eh  = (_Float16*)alloc(act * 2);
  _Float16* qh  = (_Float16*)alloc(act * 2);
  _Float16* kh  = (_Float16*)alloc(act * 2);
  _Float16* vh  = (_Float16*)alloc(act * 2);
  _Float16* oh  = (_Float16*)alloc(act * 2);
  _Float16* nh  = (_Float16*)alloc(act * 2);
  _Float16* ffh = (_Float16*)alloc((size_t)R * F * 2);
  float*    xa  = (float*)alloc(act * 4);
  float*    xb  = (float*)alloc(act * 4);

  _Float16* wh[10];
  size_t wsz[10];
  for (int i = 0; i < 8; ++i) wsz[i] = (size_t)D_MODEL * D_MODEL;
  wsz[8] = (size_t)D_MODEL * F;
  wsz[9] = (size_t)F * D_MODEL;
  for (int i = 0; i < 10; ++i) wh[i] = (_Float16*)alloc(wsz[i] * 2);

  auto conv = [&](const float* src, _Float16* dst, size_t n) {
    int blocks = (int)((n + 255) / 256);
    if (blocks > 4096) blocks = 4096;
    f32_to_f16_kernel<<<blocks, 256, 0, stream>>>(src, dst, (int)n);
  };
  auto gemm = [&](const _Float16* A, const _Float16* W, const float* bias,
                  const float* resid, float* Cf, _Float16* Ch,
                  int M, int N, int K, int relu) {
    dim3 g(N / 64, M / 128);
    gemm_wmma_kernel<<<g, 256, 0, stream>>>(A, W, bias, resid, Cf, Ch, M, N, K, relu);
  };
  auto attn = [&](const _Float16* Qp, const _Float16* Kp, const _Float16* Vp,
                  _Float16* Op, const int* m, int mB, int mQ) {
    int blocks = B * HEADS * (L / 64);
    attn_wmma_kernel<<<blocks, 128, 0, stream>>>(Qp, Kp, Vp, Op, m, L, mB, mQ, 0.125f);
  };
  auto ln = [&](const float* X, const float* a, const float* be,
                float* Yf, _Float16* Yh) {
    layernorm_kernel<<<(R + 7) / 8, 256, 0, stream>>>(X, a, be, Yf, Yh, R);
  };

  // --- precision conversion of activations + weights ---
  conv(x, xh, act);
  conv(e, eh, act);
  const int widx[10] = {4, 6, 8, 10, 12, 14, 16, 18, 20, 22};
  for (int i = 0; i < 10; ++i) conv((const float*)d_in[widx[i]], wh[i], wsz[i]);

  const float* bq1 = (const float*)d_in[5];
  const float* bk1 = (const float*)d_in[7];
  const float* bv1 = (const float*)d_in[9];
  const float* bo1 = (const float*)d_in[11];
  const float* bq2 = (const float*)d_in[13];
  const float* bk2 = (const float*)d_in[15];
  const float* bv2 = (const float*)d_in[17];
  const float* bo2 = (const float*)d_in[19];
  const float* bf1 = (const float*)d_in[21];
  const float* bf2 = (const float*)d_in[23];
  const float* a1  = (const float*)d_in[24];
  const float* be1 = (const float*)d_in[25];
  const float* a2  = (const float*)d_in[26];
  const float* be2 = (const float*)d_in[27];
  const float* a3  = (const float*)d_in[28];
  const float* be3 = (const float*)d_in[29];

  // --- self-attention block ---
  gemm(xh, wh[0], bq1, nullptr, nullptr, qh, R, D_MODEL, D_MODEL, 0);
  gemm(xh, wh[1], bk1, nullptr, nullptr, kh, R, D_MODEL, D_MODEL, 0);
  gemm(xh, wh[2], bv1, nullptr, nullptr, vh, R, D_MODEL, D_MODEL, 0);
  attn(qh, kh, vh, oh, trg_mask, L * L, L);
  gemm(oh, wh[3], bo1, x, xa, nullptr, R, D_MODEL, D_MODEL, 0);
  ln(xa, a1, be1, xb, nh);

  // --- cross-attention block ---
  gemm(nh, wh[4], bq2, nullptr, nullptr, qh, R, D_MODEL, D_MODEL, 0);
  gemm(eh, wh[5], bk2, nullptr, nullptr, kh, R, D_MODEL, D_MODEL, 0);
  gemm(eh, wh[6], bv2, nullptr, nullptr, vh, R, D_MODEL, D_MODEL, 0);
  attn(qh, kh, vh, oh, src_mask, L, 0);
  gemm(oh, wh[7], bo2, xb, xa, nullptr, R, D_MODEL, D_MODEL, 0);
  ln(xa, a2, be2, xb, nh);

  // --- feed-forward block ---
  gemm(nh, wh[8], bf1, nullptr, nullptr, ffh, R, F, D_MODEL, 1);
  gemm(ffh, wh[9], bf2, xb, xa, nullptr, R, D_MODEL, F, 0);
  ln(xa, a3, be3, (float*)d_out, nullptr);
}
